// MSDeformAttnPixelDecoder_80264348827680
// MI455X (gfx1250) — matrix operand order repaired and checked by reading, not compiled, not run on hardware
//
#include <hip/hip_runtime.h>
#include <hip/hip_bf16.h>
#include <math.h>

// MSDeformAttn pixel-decoder encoder for MI455X (gfx1250, wave32).
// GEMMs via v_wmma_f32_16x16x32_f16 with triple-buffered async-DMA tile
// staging (global_load_async_to_lds_b128, ASYNCcnt pipelined, one barrier
// per K-step, branch-free steady-state loop).

#define D_      256
#define HEADS_  8
#define DH_     32
#define LVLS_   4
#define PTS_    4
#define LAYERS_ 6
#define FFN_    1024
#define LEN_    5440
#define B_      4
#define M_      (B_*LEN_)   // 21760 = 170 * 128

typedef __attribute__((ext_vector_type(16))) _Float16 v16h;
typedef __attribute__((ext_vector_type(8)))  _Float16 v8h;
typedef __attribute__((ext_vector_type(8)))  float    v8f;

// ---------------------------------------------------------------------------
// Pack pyramid levels: x[b,q,d] = src_lvl[b,d,pix]; pos += level_embed[lvl].
// ---------------------------------------------------------------------------
__global__ __launch_bounds__(256) void pack_kernel(
    const float* __restrict__ s0, const float* __restrict__ s1,
    const float* __restrict__ s2, const float* __restrict__ s3,
    const float* __restrict__ p0, const float* __restrict__ p1,
    const float* __restrict__ p2, const float* __restrict__ p3,
    const float* __restrict__ lev,
    float* __restrict__ x, float* __restrict__ pos)
{
    size_t idx = (size_t)blockIdx.x * 256 + threadIdx.x;
    if (idx >= (size_t)M_ * D_) return;
    int d = (int)(idx & 255);
    size_t t = idx >> 8;
    int q = (int)(t % LEN_);
    int b = (int)(t / LEN_);
    int lvl, pix, hw;
    const float *s, *p;
    if (q < 4096)      { lvl = 0; pix = q;        hw = 4096; s = s0; p = p0; }
    else if (q < 5120) { lvl = 1; pix = q - 4096; hw = 1024; s = s1; p = p1; }
    else if (q < 5376) { lvl = 2; pix = q - 5120; hw = 256;  s = s2; p = p2; }
    else               { lvl = 3; pix = q - 5376; hw = 64;   s = s3; p = p3; }
    size_t gi = ((size_t)b * D_ + d) * hw + pix;
    x[idx]   = s[gi];
    pos[idx] = p[gi] + lev[lvl * D_ + d];
}

// ---------------------------------------------------------------------------
// Elementwise converts.
// ---------------------------------------------------------------------------
__global__ __launch_bounds__(256) void f32_to_f16_kernel(
    const float* __restrict__ s, _Float16* __restrict__ d, size_t n)
{
    size_t i = (size_t)blockIdx.x * 256 + threadIdx.x;
    if (i < n) d[i] = (_Float16)s[i];
}

// Transposing weight convert: w (K x N) f32 -> wt (N x K) f16.
__global__ __launch_bounds__(256) void wt_f16_kernel(
    const float* __restrict__ w, _Float16* __restrict__ wt, int K, int N)
{
    size_t i = (size_t)blockIdx.x * 256 + threadIdx.x;
    if (i >= (size_t)K * N) return;
    int n = (int)(i % N);
    int k = (int)(i / N);
    wt[(size_t)n * K + k] = (_Float16)w[i];
}

__global__ __launch_bounds__(256) void qx_kernel(
    const float* __restrict__ x, const float* __restrict__ p,
    _Float16* __restrict__ q16, _Float16* __restrict__ x16, size_t n)
{
    size_t i = (size_t)blockIdx.x * 256 + threadIdx.x;
    if (i < n) {
        float xv = x[i];
        q16[i] = (_Float16)(xv + p[i]);
        x16[i] = (_Float16)xv;
    }
}

// ---------------------------------------------------------------------------
// WMMA GEMM: C[M,N] = A[M,K](f16,row-major) * Wt[N,K](f16,transposed) + bias.
// Block = 256 threads = 8 waves; block tile 128(M) x 64(N); K step 32.
// Triple-buffered LDS; tile i+1 DMA'd (global_load_async_to_lds_b128) while
// tile i feeds the WMMAs; s_wait_asynccnt 3 retires only the older group.
// Drain iteration peeled so the steady-state loop is branch-free.
// relu_f16 == 0: f32 store to Cf.  relu_f16 == 1: relu + f16 store to Ch.
// ---------------------------------------------------------------------------
__global__ __launch_bounds__(256) void gemm_wmma_kernel(
    const _Float16* __restrict__ A, const _Float16* __restrict__ Wt,
    const float* __restrict__ bias, float* __restrict__ Cf,
    _Float16* __restrict__ Ch, int N, int K, int relu_f16)
{
    __shared__ __align__(32) _Float16 lA[3][128 * 32];  // 3 x 8 KB, (m,k)
    __shared__ __align__(32) _Float16 lB[3][64 * 32];   // 3 x 4 KB, (n,k)
    const int tid  = threadIdx.x;
    const int lane = tid & 31;
    const int wave = tid >> 5;
    const int hi   = lane >> 4;         // half-wave select
    const int m0   = blockIdx.x * 128;
    const int n0   = blockIdx.y * 64;
    const int mrow = wave * 16 + (lane & 15);
    const int ncol = lane & 15;

    // Per-thread staging slots (fixed for the whole K loop).
    const int ar = tid >> 1, ah = tid & 1;   // A: 128 rows x two 16-half chunks
    const int br = tid >> 2, bq = tid & 3;   // B: 64 rows x four 8-half chunks
    const unsigned ldsA0 = (unsigned)(size_t)&lA[0][ar * 32 + ah * 16];
    const unsigned ldsB0 = (unsigned)(size_t)&lB[0][br * 32 + bq * 8];
    const unsigned long long gA0 =
        (unsigned long long)(size_t)(A + (size_t)(m0 + ar) * K + ah * 16);
    const unsigned long long gB0 =
        (unsigned long long)(size_t)(Wt + (size_t)(n0 + br) * K + bq * 8);

    // Async DMA of one 128x32 A tile chunk (32 B) + one 64x32 Wt chunk (16 B)
    // per thread into LDS buffer `buf`.  INST_OFFSET applies to both LDS and
    // memory addresses, so the second A b128 is just offset:16.
    auto issue = [&](int buf, int k0) {
        unsigned la = ldsA0 + (unsigned)buf * (128 * 32 * 2);
        unsigned lb = ldsB0 + (unsigned)buf * (64 * 32 * 2);
        unsigned long long ga = gA0 + (unsigned long long)k0 * 2;
        unsigned long long gb = gB0 + (unsigned long long)k0 * 2;
        asm volatile(
            "global_load_async_to_lds_b128 %0, %2, off\n\t"
            "global_load_async_to_lds_b128 %0, %2, off offset:16\n\t"
            "global_load_async_to_lds_b128 %1, %3, off"
            :: "v"(la), "v"(lb), "v"(ga), "v"(gb)
            : "memory");
    };

    v8f acc[4] = {};

    // One K-step of WMMAs from LDS buffer `buf`.
    auto compute = [&](int buf) {
        const _Float16* la = &lA[0][0] + (size_t)buf * (128 * 32);
        const _Float16* lb = &lB[0][0] + (size_t)buf * (64 * 32);
        // A fragment: lane m = lane%16,
        // halves j -> K = hi*16 + j  => one contiguous v16h read.
        v16h a = *(const v16h*)&la[mrow * 32 + hi * 16];
#pragma unroll
        for (int nt = 0; nt < 4; ++nt) {
            // B fragment: lane n = lane%16, halves j -> K = hi*16 + j.
            // (n,k) LDS layout makes the 16 halves contiguous: one v16h read.
            v16h b = *(const v16h*)&lb[(nt * 16 + ncol) * 32 + hi * 16];
            acc[nt] = __builtin_amdgcn_wmma_f32_16x16x32_f16(
                false, a, false, b, (short)0, acc[nt], false, false);
        }
    };

    const int NK = K >> 5;
    issue(0, 0);
    int cur = 0;
    // Steady state: prefetch tile i+1, retire tile i, compute tile i.
    for (int i = 0; i < NK - 1; ++i) {
        int nxt = cur + 1; if (nxt == 3) nxt = 0;
        issue(nxt, (i + 1) << 5);
        // 3 ops of group i+1 may stay outstanding; group i is retired.
        asm volatile("s_wait_asynccnt 3" ::: "memory");
        __syncthreads();
        compute(cur);
        cur = nxt;
        // No trailing barrier: the next issue targets buffer (i+2)%3, last
        // read two iterations ago and protected by the barrier chain.
    }
    // Drain: last tile.
    asm volatile("s_wait_asynccnt 0" ::: "memory");
    __syncthreads();
    compute(cur);

    // C layout: VGPR r, lane l -> (M = r + (l>=16)*8, N = l%16).
#pragma unroll
    for (int nt = 0; nt < 4; ++nt) {
        int gn = n0 + nt * 16 + ncol;
        float bv = bias ? bias[gn] : 0.0f;
#pragma unroll
        for (int r = 0; r < 8; ++r) {
            int gm = m0 + wave * 16 + hi * 8 + r;
            float v = acc[nt][r] + bv;
            if (relu_f16) {
                v = v > 0.0f ? v : 0.0f;
                Ch[(size_t)gm * N + gn] = (_Float16)v;
            } else {
                Cf[(size_t)gm * N + gn] = v;
            }
        }
    }
}

// ---------------------------------------------------------------------------
// Softmax over the 16 (lvl,pt) attention logits per (b,q,head).
// ---------------------------------------------------------------------------
__global__ __launch_bounds__(256) void softmax16_kernel(float* __restrict__ aw)
{
    size_t r = (size_t)blockIdx.x * 256 + threadIdx.x;
    if (r >= (size_t)M_ * HEADS_) return;
    float* p = aw + r * 16;
    float mx = -1e30f;
#pragma unroll
    for (int i = 0; i < 16; ++i) mx = fmaxf(mx, p[i]);
    float e[16];
    float s = 0.0f;
#pragma unroll
    for (int i = 0; i < 16; ++i) { e[i] = __expf(p[i] - mx); s += e[i]; }
    float inv = 1.0f / s;
#pragma unroll
    for (int i = 0; i < 16; ++i) p[i] = e[i] * inv;
}

// ---------------------------------------------------------------------------
// Deformable attention core: block = (b,q); wave = head; lane = channel (DH=32).
// value:(B,LEN,256) off:(B,LEN,256) aw:(B,LEN,128) out:(B,LEN,256)
// ---------------------------------------------------------------------------
__global__ __launch_bounds__(256) void deform_kernel(
    const float* __restrict__ value, const float* __restrict__ off,
    const float* __restrict__ aw, float* __restrict__ out)
{
    int bq   = blockIdx.x;
    int q    = bq % LEN_;
    int b    = bq / LEN_;
    int head = threadIdx.x >> 5;
    int lane = threadIdx.x & 31;

    int lw, pixq;
    if (q < 4096)      { lw = 64; pixq = q; }
    else if (q < 5120) { lw = 32; pixq = q - 4096; }
    else if (q < 5376) { lw = 16; pixq = q - 5120; }
    else               { lw = 8;  pixq = q - 5376; }
    float rx = ((pixq % lw) + 0.5f) / (float)lw;
    float ry = ((pixq / lw) + 0.5f) / (float)lw;

    size_t row = (size_t)(b * LEN_ + q);
    const float* offp = off + row * 256 + head * 32;
    const float* awp  = aw  + row * 128 + head * 16;

    float acc = 0.0f;
    int base = 0;
    const int dims[4] = {64, 32, 16, 8};
#pragma unroll
    for (int lvl = 0; lvl < 4; ++lvl) {
        int Wd = dims[lvl];
        float Wf = (float)Wd;
        const float* vbase = value + ((size_t)b * LEN_ + base) * 256 + head * 32 + lane;
#pragma unroll
        for (int pt = 0; pt < 4; ++pt) {
            float ox = offp[(lvl * 4 + pt) * 2 + 0];
            float oy = offp[(lvl * 4 + pt) * 2 + 1];
            float xx = (rx + ox / Wf) * Wf - 0.5f;
            float yy = (ry + oy / Wf) * Wf - 0.5f;
            float x0f = floorf(xx), y0f = floorf(yy);
            float dx = xx - x0f, dy = yy - y0f;
            int x0 = (int)x0f, y0 = (int)y0f;
            float a = awp[lvl * 4 + pt];
            float s = 0.0f;
            {
                int xi = x0, yi = y0;
                if (xi >= 0 && xi < Wd && yi >= 0 && yi < Wd)
                    s += (1.0f - dx) * (1.0f - dy) * vbase[(size_t)(yi * Wd + xi) * 256];
            }
            {
                int xi = x0 + 1, yi = y0;
                if (xi >= 0 && xi < Wd && yi >= 0 && yi < Wd)
                    s += dx * (1.0f - dy) * vbase[(size_t)(yi * Wd + xi) * 256];
            }
            {
                int xi = x0, yi = y0 + 1;
                if (xi >= 0 && xi < Wd && yi >= 0 && yi < Wd)
                    s += (1.0f - dx) * dy * vbase[(size_t)(yi * Wd + xi) * 256];
            }
            {
                int xi = x0 + 1, yi = y0 + 1;
                if (xi >= 0 && xi < Wd && yi >= 0 && yi < Wd)
                    s += dx * dy * vbase[(size_t)(yi * Wd + xi) * 256];
            }
            acc += a * s;
        }
        base += Wd * Wd;
    }
    out[row * 256 + head * 32 + lane] = acc;
}

// ---------------------------------------------------------------------------
// dst = LayerNorm(x + y) * g + b over D=256; one wave per row, 8 rows/block.
// ---------------------------------------------------------------------------
__global__ __launch_bounds__(256) void ln_kernel(
    const float* __restrict__ x, const float* __restrict__ y,
    const float* __restrict__ g, const float* __restrict__ bb,
    float* __restrict__ dst)
{
    int wave = threadIdx.x >> 5, lane = threadIdx.x & 31;
    size_t row = (size_t)blockIdx.x * 8 + wave;
    const float* xr = x + row * 256;
    const float* yr = y + row * 256;
    float v[8];
    float s = 0.0f;
#pragma unroll
    for (int i = 0; i < 8; ++i) { v[i] = xr[lane * 8 + i] + yr[lane * 8 + i]; s += v[i]; }
#pragma unroll
    for (int m = 16; m >= 1; m >>= 1) s += __shfl_xor(s, m, 32);
    float mean = s * (1.0f / 256.0f);
    float s2 = 0.0f;
#pragma unroll
    for (int i = 0; i < 8; ++i) { float d = v[i] - mean; s2 += d * d; }
#pragma unroll
    for (int m = 16; m >= 1; m >>= 1) s2 += __shfl_xor(s2, m, 32);
    float inv = rsqrtf(s2 * (1.0f / 256.0f) + 1e-5f);
#pragma unroll
    for (int i = 0; i < 8; ++i)
        dst[row * 256 + lane * 8 + i] =
            (v[i] - mean) * inv * g[lane * 8 + i] + bb[lane * 8 + i];
}

// ---------------------------------------------------------------------------
// Host orchestration.
// ---------------------------------------------------------------------------
extern "C" void kernel_launch(void* const* d_in, const int* in_sizes, int n_in,
                              void* d_out, int out_size, void* d_ws, size_t ws_size,
                              hipStream_t stream)
{
    (void)in_sizes; (void)n_in; (void)out_size; (void)ws_size;

    const float* src0 = (const float*)d_in[0];
    const float* src1 = (const float*)d_in[1];
    const float* src2 = (const float*)d_in[2];
    const float* src3 = (const float*)d_in[3];
    const float* pos0 = (const float*)d_in[4];
    const float* pos1 = (const float*)d_in[5];
    const float* pos2 = (const float*)d_in[6];
    const float* pos3 = (const float*)d_in[7];
    // params dict order: level_embed, Woff, boff, Wattn, battn, Wval, bval,
    // Wout, bout, ln1_g, ln1_b, W1, b1, W2, b2, ln2_g, ln2_b
    const float* lev   = (const float*)d_in[8];
    const float* Woff  = (const float*)d_in[9];
    const float* boff  = (const float*)d_in[10];
    const float* Wattn = (const float*)d_in[11];
    const float* battn = (const float*)d_in[12];
    const float* Wval  = (const float*)d_in[13];
    const float* bval  = (const float*)d_in[14];
    const float* Wout  = (const float*)d_in[15];
    const float* bout  = (const float*)d_in[16];
    const float* ln1g  = (const float*)d_in[17];
    const float* ln1b  = (const float*)d_in[18];
    const float* W1    = (const float*)d_in[19];
    const float* b1    = (const float*)d_in[20];
    const float* W2    = (const float*)d_in[21];
    const float* b2    = (const float*)d_in[22];
    const float* ln2g  = (const float*)d_in[23];
    const float* ln2b  = (const float*)d_in[24];

    char* ws = (char*)d_ws;
    size_t off = 0;
    auto alloc = [&](size_t bytes) -> char* {
        char* p = ws + off;
        off = (off + bytes + 255) & ~(size_t)255;
        return p;
    };

    float*    xf     = (float*)   alloc((size_t)M_ * D_ * 4);
    float*    posf   = (float*)   alloc((size_t)M_ * D_ * 4);
    _Float16* q16    = (_Float16*)alloc((size_t)M_ * D_ * 2);
    _Float16* x16    = (_Float16*)alloc((size_t)M_ * D_ * 2);
    float*    val    = (float*)   alloc((size_t)M_ * D_ * 4);
    float*    scr    = (float*)   alloc((size_t)M_ * D_ * 4);   // off / proj / ffn2
    float*    awb    = (float*)   alloc((size_t)M_ * 128 * 4);
    float*    att    = (float*)   alloc((size_t)M_ * D_ * 4);
    _Float16* h16    = (_Float16*)alloc((size_t)M_ * FFN_ * 2);
    _Float16* wval16 = (_Float16*)alloc((size_t)LAYERS_ * 65536 * 2);
    _Float16* woff16 = (_Float16*)alloc((size_t)LAYERS_ * 65536 * 2);
    _Float16* watt16 = (_Float16*)alloc((size_t)LAYERS_ * 32768 * 2);
    _Float16* wout16 = (_Float16*)alloc((size_t)LAYERS_ * 65536 * 2);
    _Float16* w1_16  = (_Float16*)alloc((size_t)LAYERS_ * 262144 * 2);
    _Float16* w2_16  = (_Float16*)alloc((size_t)LAYERS_ * 262144 * 2);

    const size_t total = (size_t)M_ * D_;
    const int eb = (int)((total + 255) / 256);

    pack_kernel<<<eb, 256, 0, stream>>>(src0, src1, src2, src3,
                                        pos0, pos1, pos2, pos3,
                                        lev, xf, posf);

    // Transposed f16 weights (N x K) so GEMM B tiles are k-contiguous.
    auto cvtT = [&](const float* w, _Float16* wt, int K, int N) {
        size_t n = (size_t)K * N;
        wt_f16_kernel<<<(unsigned)((n + 255) / 256), 256, 0, stream>>>(w, wt, K, N);
    };
    for (int li = 0; li < LAYERS_; ++li) {
        cvtT(Wval  + (size_t)li * 65536,  wval16 + (size_t)li * 65536,  256, 256);
        cvtT(Woff  + (size_t)li * 65536,  woff16 + (size_t)li * 65536,  256, 256);
        cvtT(Wattn + (size_t)li * 32768,  watt16 + (size_t)li * 32768,  256, 128);
        cvtT(Wout  + (size_t)li * 65536,  wout16 + (size_t)li * 65536,  256, 256);
        cvtT(W1    + (size_t)li * 262144, w1_16  + (size_t)li * 262144, 256, 1024);
        cvtT(W2    + (size_t)li * 262144, w2_16  + (size_t)li * 262144, 1024, 256);
    }

    const dim3 blk(256);
    const int MT = M_ / 128;  // 170

    for (int li = 0; li < LAYERS_; ++li) {
        qx_kernel<<<eb, blk, 0, stream>>>(xf, posf, q16, x16, total);

        // value = x @ Wval + bval
        gemm_wmma_kernel<<<dim3(MT, 4), blk, 0, stream>>>(
            x16, wval16 + (size_t)li * 65536, bval + li * 256, val, nullptr, 256, 256, 0);
        // off = q @ Woff + boff
        gemm_wmma_kernel<<<dim3(MT, 4), blk, 0, stream>>>(
            q16, woff16 + (size_t)li * 65536, boff + li * 256, scr, nullptr, 256, 256, 0);
        // aw = q @ Wattn + battn
        gemm_wmma_kernel<<<dim3(MT, 2), blk, 0, stream>>>(
            q16, watt16 + (size_t)li * 32768, battn + li * 128, awb, nullptr, 128, 256, 0);

        softmax16_kernel<<<(M_ * HEADS_ + 255) / 256, blk, 0, stream>>>(awb);
        deform_kernel<<<M_, blk, 0, stream>>>(val, scr, awb, att);

        // attn_out @ Wout + bout  (reuse q16 as f16 scratch for attn_out)
        f32_to_f16_kernel<<<eb, blk, 0, stream>>>(att, q16, total);
        gemm_wmma_kernel<<<dim3(MT, 4), blk, 0, stream>>>(
            q16, wout16 + (size_t)li * 65536, bout + li * 256, scr, nullptr, 256, 256, 0);

        // x = LN1(x + proj)
        ln_kernel<<<M_ / 8, blk, 0, stream>>>(xf, scr, ln1g + li * 256, ln1b + li * 256, xf);

        // FFN: h = relu(x @ W1 + b1) (f16), out = h @ W2 + b2
        f32_to_f16_kernel<<<eb, blk, 0, stream>>>(xf, x16, total);
        gemm_wmma_kernel<<<dim3(MT, 16), blk, 0, stream>>>(
            x16, w1_16 + (size_t)li * 262144, b1 + li * 1024, nullptr, h16, 1024, 256, 1);
        gemm_wmma_kernel<<<dim3(MT, 4), blk, 0, stream>>>(
            h16, w2_16 + (size_t)li * 262144, b2 + li * 256, scr, nullptr, 256, 1024, 0);

        // x = LN2(x + ffn); final layer writes d_out directly
        float* dst = (li == LAYERS_ - 1) ? (float*)d_out : xf;
        ln_kernel<<<M_ / 8, blk, 0, stream>>>(xf, scr, ln2g + li * 256, ln2b + li * 256, dst);
    }
}
